// VanillaRNN_38714835206517
// MI455X (gfx1250) — matrix operand order repaired and checked by reading, not compiled, not run on hardware
//
#include <hip/hip_runtime.h>
#include <hip/hip_bf16.h>
#include <math.h>

typedef __bf16 bf16;
typedef __attribute__((ext_vector_type(8)))  __bf16 v8bf;
typedef __attribute__((ext_vector_type(16))) __bf16 v16bf;
typedef __attribute__((ext_vector_type(8)))  float  v8f;
typedef __attribute__((ext_vector_type(4)))  float  v4f;

#define T_LEN 2048
#define B_TOT 512
#define NH    256
#define NC    10
#define ROWS  16      // batch rows per workgroup (WMMA M)
#define HPAD  264     // 256 + 8 bf16 pad: row stride 528B -> +4 banks/row
#define CH    64      // x staging chunk (time steps)

__device__ __forceinline__ float fast_tanh(float c) {
#if __has_builtin(__builtin_amdgcn_tanhf)
    return __builtin_amdgcn_tanhf(c);     // native V_TANH_F32 (gfx1250 TRANS op)
#else
    return tanhf(c);
#endif
}

__device__ __forceinline__ void sched_pin_loads_then_wmma() {
#if __has_builtin(__builtin_amdgcn_sched_group_barrier)
    // Group 1: all DS reads of this step (16 A-operand b128 + 2 x b128)
    __builtin_amdgcn_sched_group_barrier(0x100, 18, 0);
    // Group 2: the 8 chained WMMAs
    __builtin_amdgcn_sched_group_barrier(0x008, 8, 0);
#endif
}

__global__ __launch_bounds__(512)
void VanillaRNN_wmma_kernel(const float* __restrict__ x,
                            const float* __restrict__ W_hx,
                            const float* __restrict__ W_hh,
                            const float* __restrict__ b_h,
                            const float* __restrict__ W_ph,
                            const float* __restrict__ b_p,
                            float* __restrict__ out)
{
    __shared__ __align__(16) bf16  h_lds[2][ROWS][HPAD];
    __shared__ __align__(16) float x_lds[CH][ROWS];

    const int tid  = threadIdx.x;
    const int wave = tid >> 5;
    const int lane = tid & 31;
    const int half = lane >> 4;          // 0: lanes 0-15, 1: lanes 16-31
    const int l16  = lane & 15;
    const int row_base = blockIdx.x * ROWS;
    const int ncol = wave * 16 + l16;    // this lane's output column (N)

    // ---- zero both h buffers (h0 = 0) ----
    for (int i = tid; i < 2 * ROWS * HPAD; i += (int)blockDim.x)
        ((bf16*)h_lds)[i] = (bf16)0.0f;

    // ---- per-lane constants for the elementwise epilogue ----
    const float whx_n = W_hx[ncol];
    const float bh_n  = b_h[ncol];

    // ---- preload this wave's B-operand (W_hh columns) into registers ----
    // B 32x16 bf16 layout (wave32): lanes 0-15 hold K={0..7,16..23} of column N=lane,
    // lanes 16-31 hold K={8..15,24..31}; element e of v16bf = K = klo_b + e + (e>=8 ? 8 : 0).
    v16bf Bm[8];
    const int klo_b = half * 8;
    for (int kb = 0; kb < 8; ++kb) {
        v16bf b;
        #pragma unroll
        for (int e = 0; e < 16; ++e)
            b[e] = (bf16)W_hh[(size_t)(kb * 32 + klo_b + e + ((e & 8) ? 8 : 0)) * NH + ncol];
        Bm[kb] = b;
    }

    __syncthreads();

    // A 16x32 bf16 layout: lane holds row M=lane%16;
    // lanes 0-15: K = {0..7, 16..23}; lanes 16-31: K = {8..15, 24..31}.
    const int klo_a = half * 8;
    const int rowA  = l16;
    const int moff  = half * 8;          // C layout: lane holds rows moff..moff+7
    int cur = 0;

    for (int t = 0; t < T_LEN; ++t) {
        const int tc = t & (CH - 1);
        if (tc == 0) {
            // stage x[row_base..+15][t..t+CH-1], transposed, coalesced along time
            for (int i = tid; i < ROWS * CH; i += (int)blockDim.x) {
                int m  = i / CH;
                int cc = i % CH;
                x_lds[cc][m] = x[(size_t)(row_base + m) * T_LEN + t + cc];
            }
            __syncthreads();
        }

        // ---- issue ALL independent LDS loads for this step up front ----
        v8bf alo[8], ahi[8];
        #pragma unroll
        for (int kb = 0; kb < 8; ++kb) {
            const bf16* ap = &h_lds[cur][rowA][kb * 32 + klo_a];
            alo[kb] = *(const v8bf*)ap;          // ds_load_b128
            ahi[kb] = *(const v8bf*)(ap + 16);   // ds_load_b128
        }
        v4f xv0 = *(const v4f*)&x_lds[tc][moff];
        v4f xv1 = *(const v4f*)&x_lds[tc][moff + 4];

        // ---- acc = h_t @ W_hh  (one 16x16 f32 tile per wave, K=256) ----
        v8f acc = {0.f,0.f,0.f,0.f,0.f,0.f,0.f,0.f};
        #pragma unroll
        for (int kb = 0; kb < 8; ++kb) {
            v16bf a = __builtin_shufflevector(alo[kb], ahi[kb],
                        0,1,2,3,4,5,6,7,8,9,10,11,12,13,14,15);
            acc = __builtin_amdgcn_wmma_f32_16x16x32_bf16(
                      false, a, false, Bm[kb], (short)0, acc, false, false);
        }
        // Pin the schedule: all 18 DS reads first, then the 8 WMMAs, so the
        // loads pipeline (incremental s_wait_dscnt) instead of serializing.
        sched_pin_loads_then_wmma();

        // ---- epilogue: + x_t * W_hx + b_h, tanh (native), store next h ----
        #pragma unroll
        for (int r = 0; r < 8; ++r) {
            float xm = (r < 4) ? xv0[r] : xv1[r - 4];
            float c  = acc[r] + xm * whx_n + bh_n;
            float hn = fast_tanh(c);
            h_lds[cur ^ 1][moff + r][ncol] = (bf16)hn;  // ds_store_b16
        }
        __syncthreads();
        cur ^= 1;
    }

    // ---- final projection: p = h_T @ W_ph + b_p  (tiny, scalar) ----
    if (tid < ROWS * NC) {
        int m = tid / NC;
        int c = tid % NC;
        float s = b_p[c];
        for (int k = 0; k < NH; ++k)
            s += (float)h_lds[cur][m][k] * W_ph[(size_t)k * NC + c];
        out[(size_t)(row_base + m) * NC + c] = s;
    }
}

extern "C" void kernel_launch(void* const* d_in, const int* in_sizes, int n_in,
                              void* d_out, int out_size, void* d_ws, size_t ws_size,
                              hipStream_t stream) {
    (void)in_sizes; (void)n_in; (void)out_size; (void)d_ws; (void)ws_size;
    const float* x    = (const float*)d_in[0];
    const float* W_hx = (const float*)d_in[1];
    const float* W_hh = (const float*)d_in[2];
    const float* b_h  = (const float*)d_in[3];
    const float* W_ph = (const float*)d_in[4];
    const float* b_p  = (const float*)d_in[5];
    float* out = (float*)d_out;

    dim3 grid(B_TOT / ROWS);   // 32 independent batch tiles
    dim3 block(512);           // 16 waves (wave32)
    VanillaRNN_wmma_kernel<<<grid, block, 0, stream>>>(x, W_hx, W_hh, b_h,
                                                       W_ph, b_p, out);
}
